// PthArcLoss_9612136809094
// MI455X (gfx1250) — matrix operand order repaired and checked by reading, not compiled, not run on hardware
//
#include <hip/hip_runtime.h>
#include <hip/hip_bf16.h>
#include <math.h>

#define NUM_CLASSES 100000
#define EMB 512
#define BATCH 512
#define S_SCALE 64.0f
// margin constants for m = 0.5
#define COS_M 0.87758256189037271612f          // cos(0.5)
#define SIN_M 0.47942553860420300027f          // sin(0.5)
#define MM_C  0.23971276930210150014f          // sin(pi-0.5)*0.5
#define THRESH (-0.87758256189037271612f)      // cos(pi-0.5)

#define C_TILE 48
#define N_TILE 128
#define NT_CNT (BATCH / N_TILE)                 // 4 batch tiles, all live in regs
#define KLS_STRIDE 520                          // 512 + 8 halves pad (bank-conflict-free)
#define C_BLOCKS ((NUM_CLASSES + C_TILE - 1) / C_TILE)   // 2084

typedef _Float16 v2h  __attribute__((ext_vector_type(2)));
typedef _Float16 v8h  __attribute__((ext_vector_type(8)));
typedef _Float16 v16h __attribute__((ext_vector_type(16)));
typedef float    v8f  __attribute__((ext_vector_type(8)));

// ---------------------------------------------------------------------------
// Kernel 1: l2-normalize embeddings, store as f16 (row-major [BATCH][EMB])
// ---------------------------------------------------------------------------
__global__ void __launch_bounds__(256)
norm_emb_kernel(const float* __restrict__ emb, _Float16* __restrict__ eA)
{
    const int row  = blockIdx.x;
    const int tid  = threadIdx.x;
    const int lane = tid & 31;
    const int wv   = tid >> 5;

    const float* src = emb + (size_t)row * EMB;
    float v0 = src[tid];
    float v1 = src[tid + 256];
    float ss = v0 * v0 + v1 * v1;
    #pragma unroll
    for (int m = 16; m >= 1; m >>= 1) ss += __shfl_xor(ss, m, 32);

    __shared__ float red[8];
    if (lane == 0) red[wv] = ss;
    __syncthreads();
    float tot = 0.f;
    #pragma unroll
    for (int i = 0; i < 8; ++i) tot += red[i];

    float inv = 1.0f / sqrtf(tot);
    eA[(size_t)row * EMB + tid]       = (_Float16)(v0 * inv);
    eA[(size_t)row * EMB + tid + 256] = (_Float16)(v1 * inv);
}

// ---------------------------------------------------------------------------
// Kernel 2: per-C-block normalized GEMM (WMMA f16 -> f32) + exp partial sums.
// Block = 256 threads = 8 waves; wave w owns M-tile w of every batch tile.
// K-loop outermost: each B fragment is read from LDS exactly ONCE per K-step
// and reused across all 4 batch tiles (4x LDS traffic reduction vs t-outer).
// ---------------------------------------------------------------------------
__global__ void __launch_bounds__(256)
arc_gemm_partial(const float* __restrict__ W, const _Float16* __restrict__ A,
                 float* __restrict__ partials)
{
    extern __shared__ char smem[];
    _Float16* kls    = (_Float16*)smem;                              // [C_TILE][KLS_STRIDE]
    float*    rowacc = (float*)(smem + C_TILE * KLS_STRIDE * 2);     // [BATCH]

    const int tid   = threadIdx.x;
    const int lane  = tid & 31;
    const int w     = tid >> 5;            // wave 0..7
    const int cb    = blockIdx.x;
    const int mlo   = lane & 15;
    const int hhalf = (lane >> 4) & 1;     // lane half
    const int akb   = hhalf ? 8 : 0;       // A: K offset per ISA 16-bit A layout
    const int bkb   = hhalf ? 8 : 0;       // B: K offset, same as A layout

    // ---- stage C_TILE weight rows: compute norm, store normalized f16 ----
    for (int rr = 0; rr < C_TILE / 8; ++rr) {
        const int r = w * (C_TILE / 8) + rr;
        const int c = cb * C_TILE + r;
        _Float16* dst = kls + r * KLS_STRIDE;
        if (c < NUM_CLASSES) {
            const float* src = W + (size_t)c * EMB;
            float2 v[8];
            float ss = 0.f;
            #pragma unroll
            for (int i = 0; i < 8; ++i) {
                v[i] = *(const float2*)(src + 2 * lane + 64 * i);
                ss += v[i].x * v[i].x + v[i].y * v[i].y;
            }
            #pragma unroll
            for (int m = 16; m >= 1; m >>= 1) ss += __shfl_xor(ss, m, 32);
            const float inv = 1.0f / sqrtf(ss);
            #pragma unroll
            for (int i = 0; i < 8; ++i) {
                v2h hv = { (_Float16)(v[i].x * inv), (_Float16)(v[i].y * inv) };
                *(v2h*)(dst + 2 * lane + 64 * i) = hv;
            }
        } else {
            v2h zv = { (_Float16)0.f, (_Float16)0.f };
            #pragma unroll
            for (int i = 0; i < 8; ++i) *(v2h*)(dst + 2 * lane + 64 * i) = zv;
        }
    }
    __syncthreads();

    // ---- main GEMM: K outermost, B fragments hoisted, 12 WMMA per K-step ----
    v8f acc[NT_CNT][3];
    #pragma unroll
    for (int t = 0; t < NT_CNT; ++t)
        #pragma unroll
        for (int ct = 0; ct < 3; ++ct)
            #pragma unroll
            for (int j = 0; j < 8; ++j) acc[t][ct][j] = 0.f;

    const _Float16* abase = A + (size_t)(w * 16 + mlo) * EMB;

    #pragma unroll 2
    for (int k0 = 0; k0 < EMB; k0 += 32) {
        // B fragments: column = lane&15; K={0..7,16..23} / {8..15,24..31}; loaded once
        v16h bv[3];
        #pragma unroll
        for (int ct = 0; ct < 3; ++ct) {
            const _Float16* bp = kls + (ct * 16 + mlo) * KLS_STRIDE + k0 + bkb;
            v8h blo = *(const v8h*)(bp);
            v8h bhi = *(const v8h*)(bp + 16);
            #pragma unroll
            for (int j = 0; j < 8; ++j) { bv[ct][j] = blo[j]; bv[ct][8 + j] = bhi[j]; }
        }
        #pragma unroll
        for (int t = 0; t < NT_CNT; ++t) {
            // A fragment: lanes 0-15 K={0..7,16..23}, lanes 16-31 K={8..15,24..31}
            const _Float16* ap = abase + (size_t)t * N_TILE * EMB + k0 + akb;
            v8h alo = *(const v8h*)(ap);
            v8h ahi = *(const v8h*)(ap + 16);
            v16h av;
            #pragma unroll
            for (int j = 0; j < 8; ++j) { av[j] = alo[j]; av[8 + j] = ahi[j]; }
            #pragma unroll
            for (int ct = 0; ct < 3; ++ct) {
                acc[t][ct] = __builtin_amdgcn_wmma_f32_16x16x32_f16(
                    false, av, false, bv[ct], (short)0, acc[t][ct], false, false);
            }
        }
    }

    // ---- exp(logit - 64) with class masking; per-row sums over 48 classes ----
    #pragma unroll
    for (int t = 0; t < NT_CNT; ++t) {
        v8f esum;
        #pragma unroll
        for (int j = 0; j < 8; ++j) esum[j] = 0.f;
        #pragma unroll
        for (int ct = 0; ct < 3; ++ct) {
            const int cg = cb * C_TILE + ct * 16 + mlo;
            const float msk = (cg < NUM_CLASSES) ? 1.0f : 0.0f;
            #pragma unroll
            for (int j = 0; j < 8; ++j)
                esum[j] += msk * __expf(S_SCALE * acc[t][ct][j] - 64.0f);
        }
        // butterfly over the 16 N-lanes (halves stay separate for masks < 16)
        #pragma unroll
        for (int m = 1; m <= 8; m <<= 1)
            #pragma unroll
            for (int j = 0; j < 8; ++j) esum[j] += __shfl_xor(esum[j], m, 32);

        if (mlo == 0) {
            #pragma unroll
            for (int j = 0; j < 8; ++j)
                rowacc[t * N_TILE + w * 16 + hhalf * 8 + j] = esum[j]; // unique writer
        }
    }
    __syncthreads();
    #pragma unroll
    for (int i = tid; i < BATCH; i += 256)
        partials[(size_t)cb * BATCH + i] = rowacc[i];
}

// ---------------------------------------------------------------------------
// Kernel 3: target logit + ArcFace margin per batch row
// ---------------------------------------------------------------------------
__global__ void __launch_bounds__(256)
arc_target(const _Float16* __restrict__ A, const float* __restrict__ W,
           const int* __restrict__ label, float* __restrict__ zy,
           float* __restrict__ nzy)
{
    const int n    = blockIdx.x;
    const int tid  = threadIdx.x;
    const int lane = tid & 31;
    const int wv   = tid >> 5;

    const int c = label[n];
    const float*    krow = W + (size_t)c * EMB;
    const _Float16* erow = A + (size_t)n * EMB;

    float dot = 0.f, q = 0.f;
    #pragma unroll
    for (int i = 0; i < 2; ++i) {
        const int idx = tid + 256 * i;
        const float kv = krow[idx];
        dot += (float)erow[idx] * kv;
        q   += kv * kv;
    }
    #pragma unroll
    for (int m = 16; m >= 1; m >>= 1) {
        dot += __shfl_xor(dot, m, 32);
        q   += __shfl_xor(q,   m, 32);
    }
    __shared__ float sd[8], sq[8];
    if (lane == 0) { sd[wv] = dot; sq[wv] = q; }
    __syncthreads();
    if (tid == 0) {
        float D = 0.f, Q = 0.f;
        #pragma unroll
        for (int i = 0; i < 8; ++i) { D += sd[i]; Q += sq[i]; }
        const float cost = D / sqrtf(Q);                 // e already normalized
        const float zyv  = S_SCALE * cost;
        const float sint = sqrtf(fmaxf(1.0f - cost * cost, 0.0f));
        float nz         = S_SCALE * (cost * COS_M - sint * SIN_M);
        const float keep = zyv - S_SCALE * MM_C;
        nz = (cost > THRESH) ? nz : keep;                // cond = relu(cos_t - thr) != 0
        zy[n]  = zyv;
        nzy[n] = nz;
    }
}

// ---------------------------------------------------------------------------
// Kernel 4: deterministic fixed-order reduction over C-blocks + final loss
// ---------------------------------------------------------------------------
__global__ void __launch_bounds__(512)
arc_finalize(const float* __restrict__ partials, const float* __restrict__ zy,
             const float* __restrict__ nzy, float* __restrict__ out)
{
    const int tid  = threadIdx.x;
    const int lane = tid & 31;
    const int wv   = tid >> 5;

    float acc = 0.f;
    for (int cbk = 0; cbk < C_BLOCKS; ++cbk)
        acc += partials[(size_t)cbk * BATCH + tid];      // fixed order, coalesced

    const float z  = zy[tid];
    const float nz = nzy[tid];
    const float tot = acc - __expf(z - 64.0f) + __expf(nz - 64.0f);
    float nll = 64.0f + __logf(tot) - nz;                // lse - new_zy

    #pragma unroll
    for (int m = 16; m >= 1; m >>= 1) nll += __shfl_xor(nll, m, 32);
    __shared__ float red[16];
    if (lane == 0) red[wv] = nll;
    __syncthreads();
    if (tid == 0) {
        float s = 0.f;
        #pragma unroll
        for (int i = 0; i < 16; ++i) s += red[i];
        out[0] = s * (1.0f / (float)BATCH);              // GRAD_SCALE = 1
    }
}

// ---------------------------------------------------------------------------
extern "C" void kernel_launch(void* const* d_in, const int* in_sizes, int n_in,
                              void* d_out, int out_size, void* d_ws, size_t ws_size,
                              hipStream_t stream)
{
    const float* emb = (const float*)d_in[0];   // [512][512] f32
    const float* W   = (const float*)d_in[1];   // [100000][512] f32
    const int*   lab = (const int*)d_in[2];     // [512] i32
    float* out = (float*)d_out;

    char* ws = (char*)d_ws;
    _Float16* eA      = (_Float16*)ws;                                       // 512 KB
    float*    parts   = (float*)(ws + (size_t)BATCH * EMB * 2);              // C_BLOCKS*512 f32
    float*    zy      = (float*)(ws + (size_t)BATCH * EMB * 2
                                    + (size_t)C_BLOCKS * BATCH * 4);
    float*    nzy     = zy + BATCH;

    const size_t shmem = (size_t)C_TILE * KLS_STRIDE * 2 + (size_t)BATCH * 4; // 51968 B

    norm_emb_kernel<<<BATCH, 256, 0, stream>>>(emb, eA);
    arc_gemm_partial<<<C_BLOCKS, 256, shmem, stream>>>(W, eA, parts);
    arc_target<<<BATCH, 256, 0, stream>>>(eA, W, lab, zy, nzy);
    arc_finalize<<<1, 512, 0, stream>>>(parts, zy, nzy, out);
}
